// DenseGATLayer_54296976556069
// MI455X (gfx1250) — matrix operand order repaired and checked by reading, not compiled, not run on hardware
//
#include <hip/hip_runtime.h>
#include <hip/hip_bf16.h>

// ---- CDNA5 (gfx1250) wave32 WMMA types ----
typedef __attribute__((ext_vector_type(16))) __bf16 v16bf;
typedef __attribute__((ext_vector_type(8)))  float  v8f;

#define GAT_N    4096
#define GAT_K    256    // IN_DIM
#define GAT_C    256    // H*D
#define GAT_H    4
#define GAT_D    64
#define NEG_SLOPE 0.2f
#define NEG_BIG  (-1.0e30f)

// ---------------------------------------------------------------------
// Async global->LDS 16B copy: CDNA5 GLOBAL_LOAD_ASYNC_TO_LDS_B128
// (tracked by ASYNCcnt). The clang builtin wants pointers in the
// unspellable __device__/__shared__ language ASes, so emit the ISA
// instruction via inline asm (bridge-doc recommended portable path):
//   global_load_async_to_lds_b128 vdst(lds byte offset), v[addr], off
// ---------------------------------------------------------------------
__device__ __forceinline__ void async_cp16(void* lds_dst, const void* gsrc) {
  const unsigned loff =
      (unsigned)(uintptr_t)(__attribute__((address_space(3))) char*)lds_dst;
  const unsigned long long ga = (unsigned long long)(uintptr_t)gsrc;
  asm volatile("global_load_async_to_lds_b128 %0, %1, off"
               :: "v"(loff), "v"(ga) : "memory");
}

#if __has_builtin(__builtin_amdgcn_s_wait_asynccnt)
#define WAIT_ASYNC0() __builtin_amdgcn_s_wait_asynccnt(0)
#else
#define WAIT_ASYNC0() asm volatile("s_wait_asynccnt 0x0" ::: "memory")
#endif

// =====================================================================
// Kernel 1: Wh = h @ W  (4096x256 @ 256x256), bf16 WMMA, f32 accumulate.
// One wave per 16x16 output tile; 8 waves per block.
// Also emits a bf16 copy of Wh (B operand of the alpha@Wh GEMM).
// =====================================================================
__global__ __launch_bounds__(256) void gat_wh_kernel(
    const float* __restrict__ h, const float* __restrict__ W,
    float* __restrict__ Wh, __bf16* __restrict__ Whb) {
  const int lane = threadIdx.x & 31;
  const int wave = threadIdx.x >> 5;
  const int tile = blockIdx.x * 8 + wave;          // 4096 tiles total
  const int i0   = (tile >> 4) * 16;               // row tile
  const int n0   = (tile & 15) * 16;               // col tile
  const int half = lane >> 4;
  const int ml   = lane & 15;

  v8f c = {};
  for (int k0 = 0; k0 < GAT_K; k0 += 32) {
    // A 16x32 bf16: lane<16 -> M=lane, elems 0..7=K0..7, 8..15=K16..23
    //               lane>=16 -> M=lane-16, elems 0..7=K8..15, 8..15=K24..31
    v16bf a;
    const float* ap = h + (size_t)(i0 + ml) * GAT_K + k0 + half * 8;
#pragma unroll
    for (int t = 0; t < 8; ++t) {
      a[t]     = (__bf16)ap[t];
      a[t + 8] = (__bf16)ap[t + 16];
    }
    // B 32x16 bf16: lane<16 -> col=lane, K=0..7,16..23 ; lane>=16 -> col=lane-16, K=8..15,24..31
    v16bf b;
    const float* bp = W + (size_t)(k0 + half * 8) * GAT_C + n0 + ml;
#pragma unroll
    for (int t = 0; t < 16; ++t) b[t] = (__bf16)bp[((t < 8) ? t : t + 8) * GAT_C];

    c = __builtin_amdgcn_wmma_f32_16x16x32_bf16(false, a, false, b,
                                                (short)0, c, false, false);
  }
  // C: VGPR r -> rows r (lanes 0-15) / r+8 (lanes 16-31), col = lane&15
#pragma unroll
  for (int r = 0; r < 8; ++r) {
    const int row = i0 + r + half * 8;
    const int col = n0 + ml;
    Wh [(size_t)row * GAT_C + col] = c[r];
    Whb[(size_t)row * GAT_C + col] = (__bf16)c[r];
  }
}

// =====================================================================
// Kernel 2: e_i[n,h] = Wh[n,h,:]·a_i[h,:], e_j likewise. 1 wave/(n,head).
// =====================================================================
__global__ __launch_bounds__(256) void gat_e_kernel(
    const float* __restrict__ Wh, const float* __restrict__ a,
    float* __restrict__ ei, float* __restrict__ ej) {
  const int lane = threadIdx.x & 31;
  const int wave = threadIdx.x >> 5;
  const int wg   = blockIdx.x * 8 + wave;          // N*H waves
  const int n    = wg >> 2;
  const int hd   = wg & 3;
  const float* whp = Wh + (size_t)n * GAT_C + hd * GAT_D;
  const float* ap  = a + hd * (2 * GAT_D);
  float si = whp[lane] * ap[lane]      + whp[lane + 32] * ap[lane + 32];
  float sj = whp[lane] * ap[64 + lane] + whp[lane + 32] * ap[96 + lane];
#pragma unroll
  for (int o = 16; o; o >>= 1) {
    si += __shfl_xor(si, o, 32);
    sj += __shfl_xor(sj, o, 32);
  }
  if (lane == 0) { ei[n * GAT_H + hd] = si; ej[n * GAT_H + hd] = sj; }
}

// =====================================================================
// Kernel 3: fused masked softmax + alpha@Wh (flash-style).
// Block = 16-row i-tile, 16 waves = (4 heads x 4 d-slices).
// Per 32-wide j-chunk, the block stages Whb (16 KB contiguous), the
// 16x32 adj tile and the e_j slab into LDS with double-buffered
// GLOBAL_LOAD_ASYNC_TO_LDS_B128 copies (ASYNCcnt), overlapping the
// next chunk's copy with this chunk's exp/WMMA work.
// =====================================================================

// LDS buffer layout (bytes), per slot:
#define WHB_STRIDE 264                    // bf16 elems/row (512B data + 16B pad)
#define ADJ_STRIDE 36                     // floats/row (128B data + 16B pad)
#define OFF_WHB 0                         // 32 * 528  = 16896
#define OFF_ADJ 16896                     // 16 * 144  =  2304
#define OFF_EJ  19200                     // 32 * 16   =   512
#define SLOT_BYTES 19712

__device__ __forceinline__ void issue_tile_copies(
    char* slot, int tid, int i0, int jt,
    const float* __restrict__ adj, const __bf16* __restrict__ Whb,
    const float* __restrict__ ej) {
  // Whb slab: rows jt..jt+31 are one contiguous 16 KB block (256 cols).
  // 1024 x 16B chunks, 2 per thread; LDS rows padded to 528B.
  const char* wsrc = (const char*)(Whb + (size_t)jt * GAT_C);
#pragma unroll
  for (int q = 0; q < 2; ++q) {
    const int c   = tid + q * 512;        // 0..1023
    const int row = c >> 5;               // 32 chunks per 512B row
    const int sub = c & 31;
    async_cp16(slot + OFF_WHB + row * 528 + sub * 16, wsrc + c * 16);
  }
  if (tid < 128) {                        // adj tile: 16 rows x 128B
    const int r   = tid >> 3;
    const int sub = tid & 7;
    async_cp16(slot + OFF_ADJ + r * 144 + sub * 16,
               adj + (size_t)(i0 + r) * GAT_N + jt + sub * 4);
  } else if (tid < 160) {                 // e_j slab: 512B contiguous
    const int c = tid - 128;
    async_cp16(slot + OFF_EJ + c * 16, ej + (size_t)jt * GAT_H + c * 4);
  }
}

__global__ __launch_bounds__(512) void gat_attn_kernel(
    const float* __restrict__ adj, const float* __restrict__ Wh,
    const __bf16* __restrict__ Whb, const float* __restrict__ ei,
    const float* __restrict__ ej, float* __restrict__ out) {
  __shared__ __align__(16) char smem[2][SLOT_BYTES];

  const int tid  = threadIdx.x;
  const int lane = tid & 31;
  const int wave = tid >> 5;
  const int hd   = wave >> 2;                       // head 0..3
  const int ds   = wave & 3;                        // 16-wide d slice 0..3
  const int i0   = blockIdx.x * 16;
  const int half = lane >> 4;
  const int mrow = lane & 15;                       // row within tile (= M)
  const int col  = hd * GAT_D + ds * 16 + mrow;     // global out column (= N)

  const float ei_r = ei[(size_t)(i0 + mrow) * GAT_H + hd];

  v8f   acc = {};
  float m = NEG_BIG, l = 0.0f;

  issue_tile_copies(smem[0], tid, i0, 0, adj, Whb, ej);

  int cur = 0;
  for (int jt = 0; jt < GAT_N; jt += 32, cur ^= 1) {
    WAIT_ASYNC0();                        // my copies for `cur` landed
    __syncthreads();                      // everyone's copies landed; prev slot free
    if (jt + 32 < GAT_N)                  // overlap next copy with compute
      issue_tile_copies(smem[cur ^ 1], tid, i0, jt + 32, adj, Whb, ej);

    const __bf16* swhb = (const __bf16*)(smem[cur] + OFF_WHB);
    const float*  sadj = (const float*) (smem[cur] + OFF_ADJ);
    const float*  sej  = (const float*) (smem[cur] + OFF_EJ);

    // logits for this lane's row, built directly in WMMA A-matrix K order
    float sv[16];
#pragma unroll
    for (int t = 0; t < 16; ++t) {
      const int k  = half * 8 + ((t < 8) ? t : t + 8);
      const float av = sadj[mrow * ADJ_STRIDE + k];
      float e = ei_r + sej[k * GAT_H + hd];
      e = (e > 0.0f ? e : NEG_SLOPE * e) + av;
      sv[t] = (av == 0.0f) ? NEG_BIG : e;           // adj==0 -> masked
    }
    // online softmax: row max across both lane-halves
    float rmax = sv[0];
#pragma unroll
    for (int t = 1; t < 16; ++t) rmax = fmaxf(rmax, sv[t]);
    rmax = fmaxf(rmax, __shfl_xor(rmax, 16, 32));
    const float mn = fmaxf(m, rmax);
    const float f  = __expf(m - mn);                // ==1 while all masked

    float rs = 0.0f;
    v16bf pa;
#pragma unroll
    for (int t = 0; t < 16; ++t) {
      const float p = (sv[t] <= -1.0e29f) ? 0.0f : __expf(sv[t] - mn);
      rs += p;
      pa[t] = (__bf16)p;
    }
    rs += __shfl_xor(rs, 16, 32);
    l = l * f + rs;
    m = mn;

    // rescale accumulator: C VGPR r holds rows r / r+8
#pragma unroll
    for (int r = 0; r < 8; ++r) {
      const float f0 = __shfl(f, r, 32);
      const float f1 = __shfl(f, r + 8, 32);
      acc[r] *= (half ? f1 : f0);
    }

    // B 32x16 tile from LDS: lane<16 -> K rows 0..7,16..23, lane>=16 -> 8..15,24..31
    v16bf pb;
#pragma unroll
    for (int t = 0; t < 16; ++t)
      pb[t] = swhb[(half * 8 + ((t < 8) ? t : t + 8)) * WHB_STRIDE + col];

    acc = __builtin_amdgcn_wmma_f32_16x16x32_bf16(false, pa, false, pb,
                                                  (short)0, acc, false, false);
  }

  // normalize, all-masked self-loop fallback, ELU, store
#pragma unroll
  for (int r = 0; r < 8; ++r) {
    const int   row = r + half * 8;
    const float l0  = __shfl(l, r, 32);
    const float l1  = __shfl(l, r + 8, 32);
    const float lr  = half ? l1 : l0;
    const int   gr  = i0 + row;
    float v = (lr > 0.0f) ? acc[r] / lr
                          : Wh[(size_t)gr * GAT_C + col];  // softmax -> delta at self
    out[(size_t)gr * GAT_C + col] = (v > 0.0f) ? v : (__expf(v) - 1.0f);
  }
}

// =====================================================================
// Launcher. Workspace layout (bytes):
//   [0, 4MiB)        Wh   fp32  4096x256
//   [4MiB, 6MiB)     Whb  bf16  4096x256
//   [6MiB, +64KiB)   ei   fp32  4096x4
//   [.. , +64KiB)    ej   fp32  4096x4
// =====================================================================
extern "C" void kernel_launch(void* const* d_in, const int* in_sizes, int n_in,
                              void* d_out, int out_size, void* d_ws, size_t ws_size,
                              hipStream_t stream) {
  const float* h   = (const float*)d_in[0];
  const float* adj = (const float*)d_in[1];
  const float* W   = (const float*)d_in[2];
  const float* a   = (const float*)d_in[3];
  float* out = (float*)d_out;

  char* ws = (char*)d_ws;
  float*  Wh  = (float*)ws;
  __bf16* Whb = (__bf16*)(ws + 4u * 1024u * 1024u);
  float*  ei  = (float*)(ws + 6u * 1024u * 1024u);
  float*  ej  = (float*)(ws + 6u * 1024u * 1024u + 64u * 1024u);

  // 4096 output tiles, 8 waves/block -> 512 blocks
  gat_wh_kernel<<<512, 256, 0, stream>>>(h, W, Wh, Whb);
  // N*H = 16384 waves, 8/block -> 2048 blocks
  gat_e_kernel<<<2048, 256, 0, stream>>>(Wh, a, ei, ej);
  // 256 i-tiles, 16 waves (4 heads x 4 d-slices) each
  gat_attn_kernel<<<256, 512, 0, stream>>>(adj, Wh, Whb, ei, ej, out);
}
